// ZetaLSTMResonantSimple_26259430047902
// MI455X (gfx1250) — matrix operand order repaired and checked by reading, not compiled, not run on hardware
//
#include <hip/hip_runtime.h>
#include <stdint.h>

// ---------------- problem constants ----------------
#define B_SZ   64
#define SEQ    1024
#define INPUT  256
#define HID    512
#define KTOT   768          // INPUT + HID, fused GEMM K
#define NGATES 2048         // 4*HID
#define MRES   15
#define NWG    32           // persistent workgroups (one per WGP)
#define NTHR   256          // 8 waves of 32
#define HC     16           // h-columns owned per WG (32*16 = 512)

typedef __attribute__((ext_vector_type(16))) __bf16 v16bf;
typedef __attribute__((ext_vector_type(8)))  float  v8f;

union Frag { uint4 u[2]; v16bf v; };   // 32B = 16 bf16 = one WMMA A/B operand

__device__ __forceinline__ unsigned short f2bf(float f) {
  union { float f; unsigned u; } c; c.f = f;
  unsigned r = c.u + 0x7FFFu + ((c.u >> 16) & 1u);   // round-to-nearest-even
  return (unsigned short)(r >> 16);
}
__device__ __forceinline__ float bf2f(unsigned short h) {
  union { unsigned u; float f; } c; c.u = ((unsigned)h) << 16; return c.f;
}
__device__ __forceinline__ float sigmoidf_(float x) { return 1.0f / (1.0f + expf(-x)); }

// Device-wide barrier: monotonic counter, release/acquire at agent scope.
__device__ __forceinline__ void grid_barrier(unsigned* cnt, unsigned epoch) {
  __builtin_amdgcn_fence(__ATOMIC_RELEASE, "agent");
  __syncthreads();
  if (threadIdx.x == 0) {
    __hip_atomic_fetch_add(cnt, 1u, __ATOMIC_RELAXED, __HIP_MEMORY_SCOPE_AGENT);
    const unsigned target = (unsigned)gridDim.x * epoch;
    while (__hip_atomic_load(cnt, __ATOMIC_RELAXED, __HIP_MEMORY_SCOPE_AGENT) < target)
      __builtin_amdgcn_s_sleep(2);
  }
  __syncthreads();
  __builtin_amdgcn_fence(__ATOMIC_ACQUIRE, "agent");
}

// ---------------- prologue kernels ----------------
__global__ void k_conv_x(const float* __restrict__ x, unsigned short* __restrict__ xb, int n) {
  int i = blockIdx.x * blockDim.x + threadIdx.x;
  if (i < n) xb[i] = f2bf(x[i]);
}

// Wcb[n][k] : k<256 -> W_ih[n][k], else W_hh[n][k-256]   (bf16, row-major, K=768)
__global__ void k_conv_w(const float* __restrict__ Wih, const float* __restrict__ Whh,
                         unsigned short* __restrict__ Wcb) {
  int i = blockIdx.x * blockDim.x + threadIdx.x;
  if (i >= NGATES * KTOT) return;
  int nrow = i / KTOT, k = i - nrow * KTOT;
  float v = (k < INPUT) ? Wih[nrow * INPUT + k] : Whh[nrow * HID + (k - INPUT)];
  Wcb[i] = f2bf(v);
}

__global__ void k_zw(const float* __restrict__ gammas, const float* __restrict__ phi,
                     float* __restrict__ zw) {
  int t = blockIdx.x * blockDim.x + threadIdx.x;
  if (t >= SEQ) return;
  float s = 0.f, tf = (float)t;
  for (int m = 0; m < MRES; ++m) s += phi[m] * cosf(gammas[m] * tf);
  zw[t] = s * (1.0f / (float)MRES);
}

__global__ void k_init(unsigned short* __restrict__ hbuf, float* __restrict__ g1acc,
                       unsigned* __restrict__ cnt) {
  int i = blockIdx.x * blockDim.x + threadIdx.x;
  int stride = blockDim.x * gridDim.x;
  if (i == 0) *cnt = 0u;                                   // reset barrier each launch
  for (int idx = i; idx < 2 * B_SZ * HID; idx += stride) hbuf[idx] = 0;      // h0 = 0
  for (int idx = i; idx < 2 * B_SZ * MRES; idx += stride) g1acc[idx] = 0.f;
}

// ---------------- persistent recurrent kernel ----------------
__global__ __launch_bounds__(NTHR, 1)
void k_lstm(const unsigned short* __restrict__ xb,    // [B][S][256] bf16
            const unsigned short* __restrict__ Wcb,   // [2048][768] bf16
            const float* __restrict__ zw,             // [S]
            const float* __restrict__ bih, const float* __restrict__ bhh,
            const float* __restrict__ Wg1,            // [15][512]
            const float* __restrict__ bg1,            // [15]
            const float* __restrict__ Wg2,            // [15]
            const float* __restrict__ bg2,            // [1]
            unsigned short* __restrict__ hbuf,        // [2][64][512] bf16 (double buffer)
            float* __restrict__ g1acc,                // [2][64][15] partial gate-MLP sums
            unsigned* __restrict__ cnt,               // grid barrier counter
            float* __restrict__ out)                  // outputs + h_f + c_f
{
  __shared__ float sGates[B_SZ * 65];   // this WG's 64x64 gate tile (padded rows)
  __shared__ float sC[B_SZ * 17];       // persistent cell state slice [64][16]
  __shared__ float sPart[B_SZ * 16];    // per-WG gate-MLP partials [64][15 (pad 16)]
  __shared__ float sBias[64];           // fused b_ih+b_hh for our 64 gate columns
  __shared__ float sWg1[MRES * 16];     // Wg1 slice for our 16 h-columns

  const int tid  = threadIdx.x;
  const int wg   = blockIdx.x;
  const int lane = tid & 31;
  const int wave = tid >> 5;
  const int gate = wave >> 1;    // 0..3 : i,f,g,o
  const int mpair = wave & 1;    // batch rows 0..31 or 32..63

  // -------- one-time LDS init --------
  if (tid < 64) {
    int gN = (tid >> 4) * HID + wg * HC + (tid & 15);
    sBias[tid] = bih[gN] + bhh[gN];
  }
  for (int idx = tid; idx < MRES * 16; idx += NTHR) {
    int m = idx >> 4, jj = idx & 15;
    sWg1[idx] = Wg1[m * HID + wg * HC + jj];
  }
  for (int idx = tid; idx < B_SZ * 17; idx += NTHR) sC[idx] = 0.f;
  __syncthreads();

  // -------- per-lane WMMA addressing (wave32 bf16 layout) --------
  // A 16x32 bf16: lanes 0-15 rows M=0..15 hold K 0..7 & 16..23; lanes 16-31 hold K 8..15 & 24..31
  // B 32x16 bf16: lane l -> column N=l&15, same K split as A (lanes 0-15 K 0..7 & 16..23; lanes 16-31 K 8..15 & 24..31)
  const int nloc  = lane & 15;
  const int khalf = lane >> 4;
  const int klo8  = khalf * 8;
  const int ncol  = gate * HID + wg * HC + nloc;
  const uint4* bp = (const uint4*)(Wcb + (size_t)ncol * KTOT + klo8);
  const int brow0 = mpair * 32 + nloc;
  const int brow1 = brow0 + 16;

  // phase-2/3 mapping: 4 threads per batch row, 4 h-columns each
  const int pb = tid & 63;
  const int jq = tid >> 6;

  unsigned barEpoch = 0;
  float hn[4], cn[4];

  for (int t = 0; t < SEQ; ++t) {
    const int par = t & 1;
    const unsigned short* hcur  = hbuf + (size_t)par * (B_SZ * HID);
    unsigned short*       hnext = hbuf + (size_t)(par ^ 1) * (B_SZ * HID);

    // ---- phase 1: gates = [x_t ; h] @ Wc^T  (2 WMMA tiles per wave) ----
    v8f acc0 = {}; v8f acc1 = {};
    {
      const uint4* xa0 = (const uint4*)(xb + (((size_t)brow0 << 10) + (size_t)t) * INPUT + klo8);
      const uint4* xa1 = (const uint4*)(xb + (((size_t)brow1 << 10) + (size_t)t) * INPUT + klo8);
      #pragma unroll
      for (int kc = 0; kc < 8; ++kc) {               // K = 0..255 from x_t
        Frag bf_, a0_, a1_;
        bf_.u[0] = bp[kc * 4];     bf_.u[1] = bp[kc * 4 + 2];
        a0_.u[0] = xa0[kc * 4];    a0_.u[1] = xa0[kc * 4 + 2];
        a1_.u[0] = xa1[kc * 4];    a1_.u[1] = xa1[kc * 4 + 2];
        acc0 = __builtin_amdgcn_wmma_f32_16x16x32_bf16(false, a0_.v, false, bf_.v, (short)0, acc0, false, false);
        acc1 = __builtin_amdgcn_wmma_f32_16x16x32_bf16(false, a1_.v, false, bf_.v, (short)0, acc1, false, false);
      }
      const uint4* ha0 = (const uint4*)(hcur + (size_t)brow0 * HID + klo8);
      const uint4* ha1 = (const uint4*)(hcur + (size_t)brow1 * HID + klo8);
      #pragma unroll
      for (int kc = 0; kc < 16; ++kc) {              // K = 256..767 from h_{t-1}
        Frag bf_, a0_, a1_;
        bf_.u[0] = bp[(kc + 8) * 4]; bf_.u[1] = bp[(kc + 8) * 4 + 2];
        a0_.u[0] = ha0[kc * 4];      a0_.u[1] = ha0[kc * 4 + 2];
        a1_.u[0] = ha1[kc * 4];      a1_.u[1] = ha1[kc * 4 + 2];
        acc0 = __builtin_amdgcn_wmma_f32_16x16x32_bf16(false, a0_.v, false, bf_.v, (short)0, acc0, false, false);
        acc1 = __builtin_amdgcn_wmma_f32_16x16x32_bf16(false, a1_.v, false, bf_.v, (short)0, acc1, false, false);
      }
    }
    // D-tile (VGPR r: M=r lanes 0-15, M=r+8 lanes 16-31; N=lane&15) -> LDS
    {
      const int mo = (lane >> 4) * 8;
      const int cg = gate * 16 + (lane & 15);
      #pragma unroll
      for (int r = 0; r < 8; ++r) {
        sGates[(mpair * 32 +      mo + r) * 65 + cg] = acc0[r];
        sGates[(mpair * 32 + 16 + mo + r) * 65 + cg] = acc1[r];
      }
    }
    for (int idx = tid; idx < B_SZ * 16; idx += NTHR) sPart[idx] = 0.f;
    __syncthreads();

    // ---- phase 2: pointwise LSTM cell + gate-MLP partial reduction ----
    #pragma unroll
    for (int u = 0; u < 4; ++u) {
      const int jj = jq * 4 + u;
      const float iv = sGates[pb * 65 +      jj] + sBias[jj];
      const float fv = sGates[pb * 65 + 16 + jj] + sBias[16 + jj];
      const float gv = sGates[pb * 65 + 32 + jj] + sBias[32 + jj];
      const float ov = sGates[pb * 65 + 48 + jj] + sBias[48 + jj];
      const float co = sC[pb * 17 + jj];
      const float cnew = sigmoidf_(fv) * co + sigmoidf_(iv) * tanhf(gv);
      sC[pb * 17 + jj] = cnew;
      cn[u] = cnew;
      hn[u] = sigmoidf_(ov) * tanhf(cnew);
    }
    #pragma unroll
    for (int m = 0; m < MRES; ++m) {
      const float p = hn[0] * sWg1[m * 16 + jq * 4]     + hn[1] * sWg1[m * 16 + jq * 4 + 1]
                    + hn[2] * sWg1[m * 16 + jq * 4 + 2] + hn[3] * sWg1[m * 16 + jq * 4 + 3];
      __hip_atomic_fetch_add(&sPart[pb * 16 + m], p, __ATOMIC_RELAXED, __HIP_MEMORY_SCOPE_WORKGROUP);
    }
    __syncthreads();
    for (int idx = tid; idx < B_SZ * MRES; idx += NTHR) {
      const int b2 = idx / MRES, m2 = idx - b2 * MRES;
      __hip_atomic_fetch_add(&g1acc[par * (B_SZ * MRES) + idx], sPart[b2 * 16 + m2],
                             __ATOMIC_RELAXED, __HIP_MEMORY_SCOPE_AGENT);
    }
    if (wg == 0) {  // pre-zero next step's accumulator (safe: last read was before prev barrier)
      for (int idx = tid; idx < B_SZ * MRES; idx += NTHR)
        g1acc[(par ^ 1) * (B_SZ * MRES) + idx] = 0.f;
    }
    grid_barrier(cnt, ++barEpoch);

    // ---- phase 3: resonant gate + modulated output ----
    float a = 0.f;
    #pragma unroll
    for (int m = 0; m < MRES; ++m)
      a += tanhf(g1acc[par * (B_SZ * MRES) + pb * MRES + m] + bg1[m]) * Wg2[m];
    const float mod = sigmoidf_(a + bg2[0]) * zw[t];
    #pragma unroll
    for (int u = 0; u < 4; ++u) {
      const int j = wg * HC + jq * 4 + u;
      const float hp = bf2f(hcur[pb * HID + j]);
      const float ho = hn[u] + mod * hp;
      out[((size_t)pb * SEQ + t) * HID + j] = ho;
      hnext[pb * HID + j] = f2bf(ho);
      if (t == SEQ - 1) {
        out[(size_t)B_SZ * SEQ * HID + (size_t)pb * HID + j] = ho;      // h_f
        out[(size_t)B_SZ * SEQ * HID + (size_t)B_SZ * HID + (size_t)pb * HID + j] = cn[u]; // c_f
      }
    }
    grid_barrier(cnt, ++barEpoch);
  }
}

// ---------------- host launcher ----------------
extern "C" void kernel_launch(void* const* d_in, const int* in_sizes, int n_in,
                              void* d_out, int out_size, void* d_ws, size_t ws_size,
                              hipStream_t stream) {
  const float* x      = (const float*)d_in[0];
  const float* Wih    = (const float*)d_in[1];
  const float* Whh    = (const float*)d_in[2];
  const float* bih    = (const float*)d_in[3];
  const float* bhh    = (const float*)d_in[4];
  const float* Wg1    = (const float*)d_in[5];
  const float* bg1    = (const float*)d_in[6];
  const float* Wg2    = (const float*)d_in[7];
  const float* bg2    = (const float*)d_in[8];
  const float* gammas = (const float*)d_in[9];
  const float* phi    = (const float*)d_in[10];
  float* out = (float*)d_out;

  // workspace layout (16B-aligned slots), total ~35.2 MB
  char* ws = (char*)d_ws;
  unsigned*       cnt   = (unsigned*)(ws + 0);
  float*          zw    = (float*)(ws + 256);                        // 4 KB
  float*          g1acc = (float*)(ws + 4608);                       // 7.5 KB
  unsigned short* hbuf  = (unsigned short*)(ws + 12288);             // 128 KB
  unsigned short* Wcb   = (unsigned short*)(ws + 143360);            // 3 MB
  unsigned short* xb    = (unsigned short*)(ws + 3289088);           // 32 MB

  const int nx = B_SZ * SEQ * INPUT;
  hipLaunchKernelGGL(k_conv_x, dim3((nx + 255) / 256), dim3(256), 0, stream, x, xb, nx);
  hipLaunchKernelGGL(k_conv_w, dim3((NGATES * KTOT + 255) / 256), dim3(256), 0, stream, Wih, Whh, Wcb);
  hipLaunchKernelGGL(k_zw,     dim3(4), dim3(256), 0, stream, gammas, phi, zw);
  hipLaunchKernelGGL(k_init,   dim3(64), dim3(256), 0, stream, hbuf, g1acc, cnt);
  hipLaunchKernelGGL(k_lstm,   dim3(NWG), dim3(NTHR), 0, stream,
                     xb, Wcb, zw, bih, bhh, Wg1, bg1, Wg2, bg2,
                     hbuf, g1acc, cnt, out);
}
